// GAT_52776558133557
// MI455X (gfx1250) — compile-verified, matrix operand order checked by reading, not run on hardware
//
#include <hip/hip_runtime.h>
#include <hip/hip_bf16.h>

#define NP   16384
#define DF   256
#define HH   4
#define HDW  1024
#define EE   131072
#define ENE  (EE + NP)
#define PP   4
#define TT   4
#define DTW  1280
#define BBG  32
#define NPGW 512
#define KKW  10
#define FINW (DTW * KKW + 1)   // 12801
#define HIDW (FINW / 2)        // 6400
#define OUTW 4

typedef __attribute__((ext_vector_type(16))) __bf16 v16bf;
typedef __attribute__((ext_vector_type(8)))  float  v8f;

__device__ __forceinline__ unsigned short f32_bf16(float f) {
  unsigned int u = __float_as_uint(f);
  u += 0x7FFFu + ((u >> 16) & 1u);           // round-to-nearest-even
  return (unsigned short)(u >> 16);
}
__device__ __forceinline__ float lrelu(float x, float s) { return x > 0.f ? x : s * x; }

// swizzled fragment order -> source K index.
// dest position rk in [0,256): kk = K-step(32), hf = lane-half, ii = pair index, par = low/high.
// matches wave32 16-bit A/B fragment layout: VGPR i holds K = (i<4 ? 2i : 16+2(i-4)) + 8*hf (+par)
__device__ __forceinline__ int swz_k(int rk) {
  const int kk = rk >> 5, q = rk & 31, hf = q >> 4, ii = (q & 15) >> 1, par = q & 1;
  return kk * 32 + ((ii < 4) ? (2 * ii) : (16 + 2 * (ii - 4))) + 8 * hf + par;
}

__device__ __forceinline__ void atomicMaxF(float* addr, float val) {
  unsigned int* ua = (unsigned int*)addr;
  unsigned int cur = *ua;
  while (__uint_as_float(cur) < val) {
    unsigned int assumed = cur;
    cur = atomicCAS(ua, assumed, __float_as_uint(val));
    if (cur == assumed) break;
  }
}

// ---------------- bf16 WMMA GEMM: C[N,1024] = A[N,256] * B[256,1024] ----------------
// Operands pre-swizzled to fragment order: [row|col][kk*32 + hf*16 + 16-elem fragment].
// Each lane: fragment = 2x global_load_b128. Per wave: 32x128 output tile,
// 16 WMMA per K-step (A frags reused 8x, B frags 2x from registers). No LDS needed.
__global__ __launch_bounds__(256) void gat_gemm_wmma_bf16(
    const unsigned short* __restrict__ As, const unsigned short* __restrict__ Bs,
    float* __restrict__ C) {
  const int tid  = threadIdx.x;
  const int wv   = tid >> 5;
  const int lane = tid & 31;
  const int mm   = lane & 15;
  const int hf   = lane >> 4;
  const int wr   = wv & 3;                  // 4 row groups of waves
  const int wc   = wv >> 2;                 // 2 col groups of waves
  const int row0 = blockIdx.y * 128 + wr * 32;
  const int col0 = blockIdx.x * 256 + wc * 128;

  v8f acc[2][8] = {};
  for (int kk = 0; kk < 8; ++kk) {
    union FragU { v16bf v; uint4 q[2]; };
    FragU a[2];
    #pragma unroll
    for (int rt = 0; rt < 2; ++rt) {
      const unsigned short* ap =
          &As[(size_t)(row0 + rt * 16 + mm) * 256 + kk * 32 + hf * 16];
      a[rt].q[0] = *(const uint4*)ap;
      a[rt].q[1] = *(const uint4*)(ap + 8);
      if (kk < 7) __builtin_prefetch(ap + 32, 0, 0);
    }
    #pragma unroll
    for (int ct = 0; ct < 8; ++ct) {
      FragU b;
      const unsigned short* bp =
          &Bs[(size_t)(col0 + ct * 16 + mm) * 256 + kk * 32 + hf * 16];
      b.q[0] = *(const uint4*)bp;
      b.q[1] = *(const uint4*)(bp + 8);
      #pragma unroll
      for (int rt = 0; rt < 2; ++rt)
        acc[rt][ct] = __builtin_amdgcn_wmma_f32_16x16x32_bf16(
            false, a[rt].v, false, b.v, (short)0, acc[rt][ct], false, false);
    }
  }
  #pragma unroll
  for (int rt = 0; rt < 2; ++rt)
    #pragma unroll
    for (int ct = 0; ct < 8; ++ct)
      #pragma unroll
      for (int j = 0; j < 8; ++j) {         // C layout: VGPR j -> M = j + 8*half
        const int r = row0 + rt * 16 + j + 8 * hf;
        C[(size_t)r * HDW + col0 + ct * 16 + mm] = acc[rt][ct][j];
      }
}

// ---------------- attention logits: wave per (n,h), shuffle reduce ----------------
__global__ void gat_att_logits(const float* __restrict__ xp,
                               const float* __restrict__ as_, const float* __restrict__ ad_,
                               float* __restrict__ als, float* __restrict__ ald) {
  const int wid  = blockIdx.x * (blockDim.x >> 5) + (threadIdx.x >> 5);
  const int lane = threadIdx.x & 31;
  if (wid >= NP * HH) return;
  const int n = wid >> 2, h = wid & 3;
  float s = 0.f, t = 0.f;
  for (int d = lane; d < DF; d += 32) {
    const float v = xp[(size_t)n * HDW + h * DF + d];
    s += v * as_[h * DF + d];
    t += v * ad_[h * DF + d];
  }
  for (int off = 16; off > 0; off >>= 1) {
    s += __shfl_down(s, off, 32);
    t += __shfl_down(t, off, 32);
  }
  if (lane == 0) { als[wid] = s; ald[wid] = t; }
}

__global__ void gat_init_mden(float* m, float* den) {
  const int i = blockIdx.x * blockDim.x + threadIdx.x;
  if (i < NP * HH) { m[i] = -1e30f; den[i] = 0.f; }
}

__global__ void gat_edge_max(const int* __restrict__ ei, const int* __restrict__ ea, int t,
                             const float* __restrict__ als, const float* __restrict__ ald,
                             float* __restrict__ m) {
  const int e = blockIdx.x * blockDim.x + threadIdx.x;
  if (e >= ENE) return;
  int src, dst; bool mask;
  if (e < EE) { src = ei[e]; dst = ei[EE + e]; mask = (ea[e] == t) && (src != dst); }
  else        { src = dst = e - EE; mask = true; }
  if (!mask) return;
  #pragma unroll
  for (int h = 0; h < HH; ++h) {
    const float lg = lrelu(als[src * HH + h] + ald[dst * HH + h], 0.2f);
    atomicMaxF(&m[dst * HH + h], lg);
  }
}

__global__ void gat_edge_sum(const int* __restrict__ ei, const int* __restrict__ ea, int t,
                             const float* __restrict__ als, const float* __restrict__ ald,
                             const float* __restrict__ m, float* __restrict__ den,
                             float* __restrict__ exb) {
  const int e = blockIdx.x * blockDim.x + threadIdx.x;
  if (e >= ENE) return;
  int src, dst; bool mask;
  if (e < EE) { src = ei[e]; dst = ei[EE + e]; mask = (ea[e] == t) && (src != dst); }
  else        { src = dst = e - EE; mask = true; }
  #pragma unroll
  for (int h = 0; h < HH; ++h) {
    float ex = 0.f;
    if (mask) {
      const float lg = lrelu(als[src * HH + h] + ald[dst * HH + h], 0.2f);
      ex = __expf(lg - m[dst * HH + h]);
      atomicAdd(&den[dst * HH + h], ex);
    }
    exb[(size_t)e * HH + h] = ex;
  }
}

// message scatter fused with head-sum; thread = (edge, 4 feature channels)
__global__ void gat_message(const int* __restrict__ ei, const float* __restrict__ exb,
                            const float* __restrict__ den, const float* __restrict__ xp,
                            float* __restrict__ nout) {
  const long long id = (long long)blockIdx.x * blockDim.x + threadIdx.x;
  if (id >= (long long)ENE * 64) return;
  const int e  = (int)(id >> 6);
  const int d0 = ((int)id & 63) * 4;
  int src, dst;
  if (e < EE) { src = ei[e]; dst = ei[EE + e]; } else { src = dst = e - EE; }
  float w[HH]; bool any = false;
  #pragma unroll
  for (int h = 0; h < HH; ++h) {
    const float ex = exb[(size_t)e * HH + h];
    w[h] = (ex > 0.f) ? ex / den[dst * HH + h] : 0.f;
    any |= (w[h] != 0.f);
  }
  if (!any) return;
  float v[4] = {0.f, 0.f, 0.f, 0.f};
  #pragma unroll
  for (int h = 0; h < HH; ++h) {
    if (w[h] == 0.f) continue;
    const float* xr = &xp[(size_t)src * HDW + h * DF + d0];
    #pragma unroll
    for (int j = 0; j < 4; ++j) v[j] += w[h] * xr[j];
  }
  #pragma unroll
  for (int j = 0; j < 4; ++j) atomicAdd(&nout[(size_t)dst * DF + d0 + j], v[j]);
}

__global__ void gat_zero(float* p, long long cnt) {
  const long long i = (long long)blockIdx.x * blockDim.x + threadIdx.x;
  if (i < cnt) p[i] = 0.f;
}

__global__ void gat_finalize_t(const float* __restrict__ nout, const float* __restrict__ bias,
                               float* __restrict__ acc) {
  const int i = blockIdx.x * blockDim.x + threadIdx.x;
  if (i >= NP * DF) return;
  const int d = i & (DF - 1);
  const float y = nout[i] * (1.f / HH) + bias[d];
  acc[i] += lrelu(y, 0.01f);
}

__global__ void gat_pass_store(const float* __restrict__ acc, float* __restrict__ xc, int p) {
  const int i = blockIdx.x * blockDim.x + threadIdx.x;
  if (i >= NP * DF) return;
  const int n = i >> 8, d = i & 255;
  xc[(size_t)n * DTW + (p + 1) * DF + d] = acc[i] * (1.f / TT);
}

// convert current x slice to bf16 in WMMA-fragment-swizzled layout
__global__ void gat_conv_x_bf16(const float* __restrict__ xc, unsigned short* __restrict__ xb, int p) {
  const int i = blockIdx.x * blockDim.x + threadIdx.x;
  if (i >= NP * DF) return;
  const int n = i >> 8;
  const int k = swz_k(i & 255);
  xb[i] = f32_bf16(xc[(size_t)n * DTW + p * DF + k]);
}

// convert W[p,t] (K-major [256][1024]) to bf16 in per-column fragment-swizzled layout
__global__ void gat_conv_w_bf16(const float* __restrict__ W, unsigned short* __restrict__ Wb,
                                long long cnt) {
  const long long i = (long long)blockIdx.x * blockDim.x + threadIdx.x;
  if (i >= cnt) return;
  const int pt = (int)(i / (DF * HDW));
  const int r  = (int)(i % (DF * HDW));
  const int c  = r >> 8;
  const int k  = swz_k(r & 255);
  Wb[i] = f32_bf16(W[(size_t)pt * DF * HDW + (size_t)k * HDW + c]);
}

__global__ void gat_copy_x(const float* __restrict__ x, float* __restrict__ xc) {
  const int i = blockIdx.x * blockDim.x + threadIdx.x;
  if (i >= NP * DF) return;
  const int n = i >> 8, d = i & 255;
  xc[(size_t)n * DTW + d] = x[i];
}

// top-K (stable, desc) per graph by last channel — one thread per graph (tiny)
__global__ void gat_sortpool(const float* __restrict__ xc, int* __restrict__ idx) {
  const int b = threadIdx.x;
  if (b >= BBG) return;
  int chosen[KKW];
  for (int k = 0; k < KKW; ++k) {
    float bv = -3.4e38f; int bi = 0;
    for (int i = 0; i < NPGW; ++i) {
      bool tk = false;
      for (int j = 0; j < k; ++j) tk |= (chosen[j] == i);
      if (tk) continue;
      const float v = xc[((size_t)b * NPGW + i) * DTW + (DTW - 1)];
      if (v > bv) { bv = v; bi = i; }
    }
    chosen[k] = bi;
    idx[b * KKW + k] = bi;
  }
}

__global__ void gat_gather(const float* __restrict__ xc, const int* __restrict__ idx,
                           const float* __restrict__ prob, float* __restrict__ hbuf) {
  const long long id  = (long long)blockIdx.x * blockDim.x + threadIdx.x;
  const long long tot = (long long)BBG * KKW * DTW;
  if (id >= tot) return;
  const int c = (int)(id % DTW);
  const int r = (int)((id / DTW) % KKW);
  const int b = (int)(id / ((long long)DTW * KKW));
  const int node = b * NPGW + idx[b * KKW + r];
  hbuf[(size_t)b * FINW + r * DTW + c] = xc[(size_t)node * DTW + c];
  if (id < BBG) hbuf[(size_t)id * FINW + FINW - 1] = prob[id];   // problemType append
}

__global__ void gat_fc(const float* __restrict__ in, const float* __restrict__ w,
                       const float* __restrict__ bias, float* __restrict__ out,
                       int inw, int outw, int act) {
  const int id = blockIdx.x * blockDim.x + threadIdx.x;
  if (id >= BBG * outw) return;
  const int j = id % outw, b = id / outw;
  float s = bias[j];
  const float* ir = &in[(size_t)b * inw];
  for (int i = 0; i < inw; ++i) s += ir[i] * w[(size_t)i * outw + j];
  out[id] = act ? lrelu(s, 0.01f) : s;
}

extern "C" void kernel_launch(void* const* d_in, const int* in_sizes, int n_in,
                              void* d_out, int out_size, void* d_ws, size_t ws_size,
                              hipStream_t stream) {
  (void)in_sizes; (void)n_in; (void)out_size; (void)ws_size;
  const float* x    = (const float*)d_in[0];
  const float* prob = (const float*)d_in[1];
  const float* W    = (const float*)d_in[2];
  const float* as_  = (const float*)d_in[3];
  const float* ad_  = (const float*)d_in[4];
  const float* bg   = (const float*)d_in[5];
  const float* f1w  = (const float*)d_in[6];
  const float* f1b  = (const float*)d_in[7];
  const float* f2w  = (const float*)d_in[8];
  const float* f2b  = (const float*)d_in[9];
  const float* f3w  = (const float*)d_in[10];
  const float* f3b  = (const float*)d_in[11];
  const int*   ei   = (const int*)d_in[12];
  const int*   ea   = (const int*)d_in[13];
  // d_in[14] (batch) unused: graphs are contiguous equal-size blocks

  size_t off = 0;
  auto carve = [&](size_t bytes) {
    void* p = (char*)d_ws + off; off += (bytes + 255) & ~(size_t)255; return p;
  };
  float* xc  = (float*)carve((size_t)NP * DTW * 4);               // JK concat buffer
  float* xp  = (float*)carve((size_t)NP * HDW * 4);               // projected features
  unsigned short* xb = (unsigned short*)carve((size_t)NP * DF * 2);
  unsigned short* Wb = (unsigned short*)carve((size_t)PP * TT * DF * HDW * 2);
  float* als = (float*)carve((size_t)NP * HH * 4);
  float* ald = (float*)carve((size_t)NP * HH * 4);
  float* mx  = (float*)carve((size_t)NP * HH * 4);
  float* den = (float*)carve((size_t)NP * HH * 4);
  float* exb = (float*)carve((size_t)ENE * HH * 4);
  float* nout= (float*)carve((size_t)NP * DF * 4);
  float* acc = (float*)carve((size_t)NP * DF * 4);
  int*   idx = (int*)carve((size_t)BBG * KKW * 4);
  float* hb  = (float*)carve((size_t)BBG * FINW * 4);
  float* h1  = (float*)carve((size_t)BBG * HIDW * 4);
  float* h2  = (float*)carve((size_t)BBG * HIDW * 4);

  const int TB  = 256;
  const int ND  = NP * DF;
  const int ndB = (ND + TB - 1) / TB;
  const int nhB = (NP * HH + TB - 1) / TB;
  const int enB = (ENE + TB - 1) / TB;

  gat_copy_x<<<ndB, TB, 0, stream>>>(x, xc);
  const long long wcnt = (long long)PP * TT * DF * HDW;
  gat_conv_w_bf16<<<(int)((wcnt + TB - 1) / TB), TB, 0, stream>>>(W, Wb, wcnt);

  for (int p = 0; p < PP; ++p) {
    gat_conv_x_bf16<<<ndB, TB, 0, stream>>>(xc, xb, p);
    gat_zero<<<ndB, TB, 0, stream>>>(acc, (long long)ND);
    for (int t = 0; t < TT; ++t) {
      const int pti = p * TT + t;
      dim3 ggrid(HDW / 256, NP / 128);
      gat_gemm_wmma_bf16<<<ggrid, 256, 0, stream>>>(xb, Wb + (size_t)pti * DF * HDW, xp);
      gat_att_logits<<<(NP * HH) / 8, 256, 0, stream>>>(
          xp, as_ + (size_t)pti * HH * DF, ad_ + (size_t)pti * HH * DF, als, ald);
      gat_init_mden<<<nhB, TB, 0, stream>>>(mx, den);
      gat_edge_max<<<enB, TB, 0, stream>>>(ei, ea, t, als, ald, mx);
      gat_edge_sum<<<enB, TB, 0, stream>>>(ei, ea, t, als, ald, mx, den, exb);
      gat_zero<<<ndB, TB, 0, stream>>>(nout, (long long)ND);
      const long long mcnt = (long long)ENE * 64;
      gat_message<<<(int)((mcnt + TB - 1) / TB), TB, 0, stream>>>(ei, exb, den, xp, nout);
      gat_finalize_t<<<ndB, TB, 0, stream>>>(nout, bg + (size_t)pti * DF, acc);
    }
    gat_pass_store<<<ndB, TB, 0, stream>>>(acc, xc, p);
  }

  gat_sortpool<<<1, 32, 0, stream>>>(xc, idx);
  const long long gcnt = (long long)BBG * KKW * DTW;
  gat_gather<<<(int)((gcnt + TB - 1) / TB), TB, 0, stream>>>(xc, idx, prob, hb);
  gat_fc<<<(BBG * HIDW + TB - 1) / TB, TB, 0, stream>>>(hb, f1w, f1b, h1, FINW, HIDW, 1);
  gat_fc<<<(BBG * HIDW + TB - 1) / TB, TB, 0, stream>>>(h1, f2w, f2b, h2, HIDW, HIDW, 1);
  gat_fc<<<(BBG * OUTW + TB - 1) / TB, TB, 0, stream>>>(h2, f3w, f3b, (float*)d_out, HIDW, OUTW, 0);
}
